// GroupedVectorAttention_80942953660747
// MI455X (gfx1250) — matrix operand order repaired and checked by reading, not compiled, not run on hardware
//
#include <hip/hip_runtime.h>

// Grouped vector attention (Point Transformer) for MI455X / gfx1250.
// All GEMM-shaped math (q/k/v proj, pe_w2, wo) runs on v_wmma_f32_16x16x32_f16.
// q/k/v/out stored point-major (B,N,C) so neighbor gathers are contiguous.

#define BB   2
#define CC   128
#define NN   8192
#define KK   16
#define GG   8
#define EPSV 1e-5f

typedef __attribute__((ext_vector_type(16))) _Float16 v16h;
typedef __attribute__((ext_vector_type(8)))  float    v8f;

// ---------- WMMA 16x16x32 f16 fragment helpers (CDNA5 wave32 layouts) ----------
// A-matrix 16x32 (MxK): lanes 0-15 hold rows M=0..15 with K 0..7 & 16..23,
// lanes 16-31 hold the same rows with K 8..15 & 24..31 (2 halves per VGPR).
__device__ inline v16h load_a_frag(const _Float16* __restrict__ A, int lda,
                                   int m0, int k0, int lane) {
  const int m  = m0 + (lane & 15);
  const int kh = (lane >> 4) << 3;   // 0 or 8
  v16h a;
#pragma unroll
  for (int v = 0; v < 8; ++v) {
    const int kb = k0 + ((v < 4) ? 0 : 16) + kh + ((v & 3) << 1);
    a[2 * v]     = A[m * lda + kb];
    a[2 * v + 1] = A[m * lda + kb + 1];
  }
  return a;
}

// B-matrix 32x16 (KxN): lane holds column n = lane&15; lanes 0-15 K=0..7 & 16..23,
// lanes 16-31 K=8..15 & 24..31, two K per VGPR. B[k][n] == W[n][k] for row-major W
// (out-ch x in-ch), so each lane streams a contiguous row chunk of W.
__device__ inline v16h load_b_frag(const _Float16* __restrict__ W, int ldw,
                                   int k0, int n0, int lane) {
  const int n  = n0 + (lane & 15);
  const int kh = (lane >> 4) << 3;   // 0 or 8
  v16h b;
#pragma unroll
  for (int v = 0; v < 8; ++v) {
    const int kb = k0 + ((v < 4) ? 0 : 16) + kh + ((v & 3) << 1);
    b[2 * v]     = W[n * ldw + kb];
    b[2 * v + 1] = W[n * ldw + kb + 1];
  }
  return b;
}
// D/C 16x16 f32: lane l holds n = l&15; VGPR v holds m = v + 8*(l>>4).

// ---------- weight fp32 -> f16 ----------
__global__ void cvt_f16_kernel(const float* __restrict__ src,
                               _Float16* __restrict__ dst, int n) {
  int i = blockIdx.x * blockDim.x + threadIdx.x;
  if (i < n) dst[i] = (_Float16)src[i];
}

// ---------- brute-force KNN: one thread per query, sorted top-16 in regs ----------
__global__ __launch_bounds__(256) void knn_kernel(const float* __restrict__ xyz,
                                                  int* __restrict__ idx) {
  const int b = blockIdx.y;
  const int n = blockIdx.x * 256 + threadIdx.x;
  const size_t base = (size_t)b * 3 * NN;
  const float qx = xyz[base + 0 * NN + n];
  const float qy = xyz[base + 1 * NN + n];
  const float qz = xyz[base + 2 * NN + n];

  float bd[KK];
  int   bi[KK];
#pragma unroll
  for (int t = 0; t < KK; ++t) { bd[t] = 3.0e38f; bi[t] = 0; }

  __shared__ float cx[256], cy[256], cz[256];
  for (int jt = 0; jt < NN; jt += 256) {
    __syncthreads();
    cx[threadIdx.x] = xyz[base + 0 * NN + jt + threadIdx.x];
    cy[threadIdx.x] = xyz[base + 1 * NN + jt + threadIdx.x];
    cz[threadIdx.x] = xyz[base + 2 * NN + jt + threadIdx.x];
    __syncthreads();
    for (int j = 0; j < 256; ++j) {
      const float dx = qx - cx[j], dy = qy - cy[j], dz = qz - cz[j];
      const float d2 = dx * dx + dy * dy + dz * dz;
      if (d2 < bd[KK - 1]) {
        bd[KK - 1] = d2; bi[KK - 1] = jt + j;
#pragma unroll
        for (int t = KK - 1; t > 0; --t) {  // unrolled bubble keeps regs, stays sorted
          if (bd[t] < bd[t - 1]) {
            float td = bd[t]; bd[t] = bd[t - 1]; bd[t - 1] = td;
            int   ti = bi[t]; bi[t] = bi[t - 1]; bi[t - 1] = ti;
          }
        }
      }
    }
  }
#pragma unroll
  for (int t = 0; t < KK; ++t)
    idx[(size_t)b * NN * KK + (size_t)n * KK + t] = bi[t];
}

// ---------- q/k/v projection: oT(B,N,C) = [relu(bn(W @ x))]^T via WMMA ----------
__global__ __launch_bounds__(256) void proj_kernel(
    const float* __restrict__ x,            // (B,C,N)
    const _Float16* __restrict__ Wh,        // (C,C) f16, row-major out x in
    const float* __restrict__ bn_g, const float* __restrict__ bn_b,
    const float* __restrict__ bn_m, const float* __restrict__ bn_v,
    int bnrelu, float* __restrict__ oT) {   // (B,N,C)
  const int b    = blockIdx.y;
  const int n0   = blockIdx.x * 128;
  const int tid  = threadIdx.x;
  const int lane = tid & 31;
  const int wave = tid >> 5;

  __shared__ _Float16 AT[128 * 132];        // x^T tile, point-major
  for (int e = tid; e < 128 * 128; e += 256) {
    const int c = e >> 7, p = e & 127;      // coalesced along points
    AT[p * 132 + c] = (_Float16)x[(size_t)b * CC * NN + (size_t)c * NN + n0 + p];
  }
  __syncthreads();

  const int ot = wave;                      // out-channel tile 0..7
  v8f acc[8] = {};
#pragma unroll
  for (int kc = 0; kc < 4; ++kc) {
    const v16h bf = load_b_frag(Wh, CC, kc * 32, ot * 16, lane);
#pragma unroll
    for (int pt = 0; pt < 8; ++pt) {
      const v16h af = load_a_frag(AT, 132, pt * 16, kc * 32, lane);
      acc[pt] = __builtin_amdgcn_wmma_f32_16x16x32_f16(
          false, af, false, bf, (short)0, acc[pt], false, false);
    }
  }

  const int oc = ot * 16 + (lane & 15);
  float sc = 1.0f, sh = 0.0f;
  if (bnrelu) {
    sc = bn_g[oc] * rsqrtf(bn_v[oc] + EPSV);
    sh = bn_b[oc] - bn_m[oc] * sc;
  }
#pragma unroll
  for (int pt = 0; pt < 8; ++pt) {
#pragma unroll
    for (int v = 0; v < 8; ++v) {
      const int p = pt * 16 + v + ((lane >> 4) << 3);
      float r = acc[pt][v];
      if (bnrelu) r = fmaxf(r * sc + sh, 0.0f);
      oT[(size_t)b * NN * CC + (size_t)(n0 + p) * CC + oc] = r;
    }
  }
}

// ---------- fused attention: 4 query points (64 neighbor-columns) per WG ----------
__global__ __launch_bounds__(256) void attn_kernel(
    const float* __restrict__ xyz,
    const float* __restrict__ qT, const float* __restrict__ kT,
    const float* __restrict__ vT, const int* __restrict__ idx,
    const float* __restrict__ pe_w1,
    const float* __restrict__ pe_g, const float* __restrict__ pe_b,
    const float* __restrict__ pe_m, const float* __restrict__ pe_v,
    const _Float16* __restrict__ pe_w2h,
    const float* __restrict__ we_w1,
    const float* __restrict__ we_g, const float* __restrict__ we_b,
    const float* __restrict__ we_m, const float* __restrict__ we_v,
    const float* __restrict__ we_w2,
    float* __restrict__ outT) {
  const int b    = blockIdx.y;
  const int n0   = blockIdx.x * 4;
  const int tid  = threadIdx.x;
  const int lane = tid & 31;
  const int wave = tid >> 5;

  __shared__ _Float16 tTs[64 * 132];   // PE hidden, col-major over channels
  __shared__ _Float16 relh[64 * 132];  // k_nbr - q + pe
  __shared__ _Float16 vnh[64 * 132];   // v_nbr + pe
  __shared__ int   nidx[64];
  __shared__ float rp[3][64];
  __shared__ float sM[GG][64];
  __shared__ float wM[GG][64];

  if (tid < 64) {
    const int qi = tid >> 4;
    nidx[tid] = idx[(size_t)b * NN * KK + (size_t)(n0 + qi) * KK + (tid & 15)];
  }
  __syncthreads();
  if (tid < 192) {
    const int d = tid >> 6, col = tid & 63;
    const int qi = col >> 4;
    const int p  = nidx[col];
    rp[d][col] = xyz[(size_t)b * 3 * NN + (size_t)d * NN + p] -
                 xyz[(size_t)b * 3 * NN + (size_t)d * NN + n0 + qi];
  }
  __syncthreads();

  // PE hidden: relu(bn(pe_w1 @ rel_pos)), K=3 -> plain VALU
  for (int e = tid; e < 64 * CC; e += 256) {
    const int col = e >> 7, ch = e & 127;
    float t = pe_w1[ch * 3 + 0] * rp[0][col] + pe_w1[ch * 3 + 1] * rp[1][col] +
              pe_w1[ch * 3 + 2] * rp[2][col];
    const float sc = pe_g[ch] * rsqrtf(pe_v[ch] + EPSV);
    t = fmaxf(t * sc + (pe_b[ch] - pe_m[ch] * sc), 0.0f);
    tTs[col * 132 + ch] = (_Float16)t;
  }
  __syncthreads();

  // pe = pe_w2 @ t : the 8.6 GFLOP GEMM, via WMMA. wave -> out-ch tile, 4 col tiles.
  const int ot = wave;
  v8f acc[4] = {};
#pragma unroll
  for (int kc = 0; kc < 4; ++kc) {
    const v16h bf = load_b_frag(pe_w2h, CC, kc * 32, ot * 16, lane);
#pragma unroll
    for (int ct = 0; ct < 4; ++ct) {
      const v16h af = load_a_frag(tTs, 132, ct * 16, kc * 32, lane);
      acc[ct] = __builtin_amdgcn_wmma_f32_16x16x32_f16(
          false, af, false, bf, (short)0, acc[ct], false, false);
    }
  }

  // fuse gather + residual directly from the D fragments
  const int oc = ot * 16 + (lane & 15);
#pragma unroll
  for (int ct = 0; ct < 4; ++ct) {
#pragma unroll
    for (int v = 0; v < 8; ++v) {
      const int col = ct * 16 + v + ((lane >> 4) << 3);
      const int p   = nidx[col];
      const int qi  = col >> 4;
      const float pe = acc[ct][v];
      const float kv = kT[(size_t)b * NN * CC + (size_t)p * CC + oc];
      const float qv = qT[(size_t)b * NN * CC + (size_t)(n0 + qi) * CC + oc];
      const float vv = vT[(size_t)b * NN * CC + (size_t)p * CC + oc];
      relh[col * 132 + oc] = (_Float16)(kv - qv + pe);
      vnh[col * 132 + oc]  = (_Float16)(vv + pe);
    }
  }
  __syncthreads();

  // s = relu(bn(we_w1 @ rel)) : G=8 rows, dot over 128 channels
  for (int o = tid; o < GG * 64; o += 256) {
    const int g = o >> 6, col = o & 63;
    float a = 0.0f;
#pragma unroll 8
    for (int c = 0; c < CC; ++c) a += we_w1[g * CC + c] * (float)relh[col * 132 + c];
    const float sc = we_g[g] * rsqrtf(we_v[g] + EPSV);
    sM[g][col] = fmaxf(a * sc + (we_b[g] - we_m[g] * sc), 0.0f);
  }
  __syncthreads();

  // w = we_w2 @ s (8x8 tiny)
  for (int o = tid; o < GG * 64; o += 256) {
    const int h = o >> 6, col = o & 63;
    float a = 0.0f;
#pragma unroll
    for (int g = 0; g < GG; ++g) a += we_w2[h * GG + g] * sM[g][col];
    wM[h][col] = a;
  }
  __syncthreads();

  // softmax over the 16 neighbors, per (group, query)
  if (tid < GG * 4) {
    const int h = tid >> 2, qi = tid & 3;
    float mx = -3.0e38f;
#pragma unroll
    for (int j = 0; j < KK; ++j) mx = fmaxf(mx, wM[h][qi * 16 + j]);
    float ex[KK], sum = 0.0f;
#pragma unroll
    for (int j = 0; j < KK; ++j) { ex[j] = __expf(wM[h][qi * 16 + j] - mx); sum += ex[j]; }
    const float inv = 1.0f / sum;
#pragma unroll
    for (int j = 0; j < KK; ++j) wM[h][qi * 16 + j] = ex[j] * inv;
  }
  __syncthreads();

  // out[qi][c] = sum_j vn[qi*16+j][c] * w[c/16][qi*16+j]
  for (int o = tid; o < 4 * CC; o += 256) {
    const int qi = o >> 7, c = o & 127;
    const int g  = c >> 4;
    float a = 0.0f;
#pragma unroll
    for (int j = 0; j < KK; ++j)
      a += (float)vnh[(qi * 16 + j) * 132 + c] * wM[g][qi * 16 + j];
    outT[(size_t)b * NN * CC + (size_t)(n0 + qi) * CC + c] = a;  // coalesced
  }
}

// ---------- final wo projection + BN, (B,N,C) in -> (B,C,N) out ----------
__global__ __launch_bounds__(256) void out_proj_kernel(
    const float* __restrict__ inT,          // (B,N,C)
    const _Float16* __restrict__ Wh,        // wo f16
    const float* __restrict__ bn_g, const float* __restrict__ bn_b,
    const float* __restrict__ bn_m, const float* __restrict__ bn_v,
    float* __restrict__ out) {              // (B,C,N)
  const int b    = blockIdx.y;
  const int n0   = blockIdx.x * 128;
  const int tid  = threadIdx.x;
  const int lane = tid & 31;
  const int wave = tid >> 5;

  __shared__ _Float16 AT[128 * 132];
  for (int e = tid; e < 128 * 128; e += 256) {
    const int p = e >> 7, c = e & 127;      // coalesced along channels
    AT[p * 132 + c] = (_Float16)inT[(size_t)b * NN * CC + (size_t)(n0 + p) * CC + c];
  }
  __syncthreads();

  const int ot = wave;
  v8f acc[8] = {};
#pragma unroll
  for (int kc = 0; kc < 4; ++kc) {
    const v16h bf = load_b_frag(Wh, CC, kc * 32, ot * 16, lane);
#pragma unroll
    for (int pt = 0; pt < 8; ++pt) {
      const v16h af = load_a_frag(AT, 132, pt * 16, kc * 32, lane);
      acc[pt] = __builtin_amdgcn_wmma_f32_16x16x32_f16(
          false, af, false, bf, (short)0, acc[pt], false, false);
    }
  }

  const int oc = ot * 16 + (lane & 15);
  const float sc = bn_g[oc] * rsqrtf(bn_v[oc] + EPSV);
  const float sh = bn_b[oc] - bn_m[oc] * sc;
#pragma unroll
  for (int pt = 0; pt < 8; ++pt) {
#pragma unroll
    for (int v = 0; v < 8; ++v) {
      const int p = pt * 16 + v + ((lane >> 4) << 3);
      out[(size_t)b * CC * NN + (size_t)oc * NN + n0 + p] = acc[pt][v] * sc + sh;
    }
  }
}

extern "C" void kernel_launch(void* const* d_in, const int* in_sizes, int n_in,
                              void* d_out, int out_size, void* d_ws, size_t ws_size,
                              hipStream_t stream) {
  (void)in_sizes; (void)n_in; (void)out_size; (void)ws_size;

  const float* x     = (const float*)d_in[0];
  const float* xyz   = (const float*)d_in[1];
  const float* wq    = (const float*)d_in[2];
  const float* bnq_g = (const float*)d_in[3];
  const float* bnq_b = (const float*)d_in[4];
  const float* bnq_m = (const float*)d_in[5];
  const float* bnq_v = (const float*)d_in[6];
  const float* wk    = (const float*)d_in[7];
  const float* bnk_g = (const float*)d_in[8];
  const float* bnk_b = (const float*)d_in[9];
  const float* bnk_m = (const float*)d_in[10];
  const float* bnk_v = (const float*)d_in[11];
  const float* wv    = (const float*)d_in[12];
  const float* pe_w1 = (const float*)d_in[13];
  const float* pe_g  = (const float*)d_in[14];
  const float* pe_b  = (const float*)d_in[15];
  const float* pe_m  = (const float*)d_in[16];
  const float* pe_v  = (const float*)d_in[17];
  const float* pe_w2 = (const float*)d_in[18];
  const float* we_w1 = (const float*)d_in[19];
  const float* we_g  = (const float*)d_in[20];
  const float* we_b  = (const float*)d_in[21];
  const float* we_m  = (const float*)d_in[22];
  const float* we_v  = (const float*)d_in[23];
  const float* we_w2 = (const float*)d_in[24];
  const float* wo    = (const float*)d_in[25];
  const float* bno_g = (const float*)d_in[26];
  const float* bno_b = (const float*)d_in[27];
  const float* bno_m = (const float*)d_in[28];
  const float* bno_v = (const float*)d_in[29];

  // workspace layout
  char* ws = (char*)d_ws;
  size_t off = 0;
  const size_t sBNC = (size_t)BB * NN * CC * sizeof(float);   // 8 MB
  float* qT   = (float*)(ws + off); off += sBNC;
  float* kT   = (float*)(ws + off); off += sBNC;
  float* vT   = (float*)(ws + off); off += sBNC;
  float* aT   = (float*)(ws + off); off += sBNC;              // attention out (B,N,C)
  int*   idxb = (int*)(ws + off);   off += (size_t)BB * NN * KK * sizeof(int);
  const size_t sW = (size_t)CC * CC * sizeof(_Float16);
  _Float16* wqh = (_Float16*)(ws + off); off += sW;
  _Float16* wkh = (_Float16*)(ws + off); off += sW;
  _Float16* wvh = (_Float16*)(ws + off); off += sW;
  _Float16* pw2h = (_Float16*)(ws + off); off += sW;
  _Float16* woh  = (_Float16*)(ws + off); off += sW;

  const int nW = CC * CC;
  cvt_f16_kernel<<<(nW + 255) / 256, 256, 0, stream>>>(wq, wqh, nW);
  cvt_f16_kernel<<<(nW + 255) / 256, 256, 0, stream>>>(wk, wkh, nW);
  cvt_f16_kernel<<<(nW + 255) / 256, 256, 0, stream>>>(wv, wvh, nW);
  cvt_f16_kernel<<<(nW + 255) / 256, 256, 0, stream>>>(pe_w2, pw2h, nW);
  cvt_f16_kernel<<<(nW + 255) / 256, 256, 0, stream>>>(wo, woh, nW);

  knn_kernel<<<dim3(NN / 256, BB), 256, 0, stream>>>(xyz, idxb);

  proj_kernel<<<dim3(NN / 128, BB), 256, 0, stream>>>(x, wqh, bnq_g, bnq_b, bnq_m, bnq_v, 1, qT);
  proj_kernel<<<dim3(NN / 128, BB), 256, 0, stream>>>(x, wkh, bnk_g, bnk_b, bnk_m, bnk_v, 1, kT);
  proj_kernel<<<dim3(NN / 128, BB), 256, 0, stream>>>(x, wvh, bnq_g, bnq_b, bnq_m, bnq_v, 0, vT);

  attn_kernel<<<dim3(NN / 4, BB), 256, 0, stream>>>(
      xyz, qT, kT, vT, idxb, pe_w1, pe_g, pe_b, pe_m, pe_v, pw2h,
      we_w1, we_g, we_b, we_m, we_v, we_w2, aT);

  out_proj_kernel<<<dim3(NN / 128, BB), 256, 0, stream>>>(
      aT, woh, bno_g, bno_b, bno_m, bno_v, (float*)d_out);
}
